// Attention_30657476559501
// MI455X (gfx1250) — matrix operand order repaired and checked by reading, not compiled, not run on hardware
//
#include <hip/hip_runtime.h>
#include <hip/hip_bf16.h>

// MI455X (gfx1250) fused attention:
//   fp32->bf16 convert (+ one-time weight transpose), WMMA bf16 GEMMs,
//   flash-attention with online softmax (S^2 matrix never leaves the WGP),
//   async global->LDS staging (ASYNCcnt) where the toolchain supports it.
// Wave32; v_wmma_f32_16x16x32_bf16 throughout.

typedef __bf16 bf16;
typedef __attribute__((ext_vector_type(16))) __bf16 v16bf;
typedef __attribute__((ext_vector_type(8)))  __bf16 v8bf;
typedef __attribute__((ext_vector_type(8)))  float  v8f;
typedef int v4i __attribute__((vector_size(16)));   // matches async-LDS builtin proto

#define HEADS  12
#define DHEAD  64
#define SEQ    4096
#define DMODEL 768
#define BATCH  2
#define MROWS  (BATCH * SEQ)   // 8192
#define ATTN_SCALE 0.125f      // 1/sqrt(64)

#if defined(__has_builtin)
#if __has_builtin(__builtin_amdgcn_global_load_async_to_lds_b128) && \
    __has_builtin(__builtin_amdgcn_s_wait_asynccnt)
#define HAVE_ASYNC_LDS 1
#endif
#endif
#ifndef HAVE_ASYNC_LDS
#define HAVE_ASYNC_LDS 0
#endif

static __device__ __forceinline__ v8f zero8() {
  v8f z = {0.f, 0.f, 0.f, 0.f, 0.f, 0.f, 0.f, 0.f};
  return z;
}

static __device__ __forceinline__ v8f wmma_bf16(v16bf a, v16bf b, v8f c) {
  // (neg_a, A, neg_b, B, c_mod, C, reuse_a, reuse_b)
  return __builtin_amdgcn_wmma_f32_16x16x32_bf16(false, a, false, b, (short)0, c,
                                                 false, false);
}

// Build a 16-element fragment from two contiguous 8-element runs.
static __device__ __forceinline__ v16bf ld_frag(const bf16* p0, const bf16* p1) {
  v8bf x0 = *reinterpret_cast<const v8bf*>(p0);
  v8bf x1 = *reinterpret_cast<const v8bf*>(p1);
  v16bf a;
#pragma unroll
  for (int i = 0; i < 8; ++i) { a[i] = x0[i]; a[8 + i] = x1[i]; }
  return a;
}

// 16-byte global -> LDS staging copy; async (ASYNCcnt) when available.
static __device__ __forceinline__ void stage16(const bf16* g, bf16* l) {
#if HAVE_ASYNC_LDS
  __builtin_amdgcn_global_load_async_to_lds_b128(
      (__attribute__((address_space(1))) v4i*)(g),
      (__attribute__((address_space(3))) v4i*)(l), 0, 0);
#else
  *reinterpret_cast<v8bf*>(l) = *reinterpret_cast<const v8bf*>(g);
#endif
}

static __device__ __forceinline__ void stage_wait() {
#if HAVE_ASYNC_LDS
  __builtin_amdgcn_s_wait_asynccnt(0);
#endif
}

__global__ __launch_bounds__(256)
void f32_to_bf16(const float* __restrict__ in, bf16* __restrict__ out, int n) {
  int i = blockIdx.x * blockDim.x + threadIdx.x;
  if (i < n) out[i] = (bf16)in[i];
}

// Transpose + convert a 768x768 weight matrix: out[n][k] = (bf16)in[k][n].
// 32x32 LDS tiles, 256 threads (32x8, 4 rows each).
__global__ __launch_bounds__(256)
void transpose_f32_to_bf16(const float* __restrict__ in, bf16* __restrict__ out) {
  __shared__ float tile[32][33];
  const int tx = threadIdx.x & 31;
  const int ty = (threadIdx.x >> 5) * 4;
  const int k0 = blockIdx.x * 32, n0 = blockIdx.y * 32;
#pragma unroll
  for (int i = 0; i < 4; ++i)
    tile[ty + i][tx] = in[(size_t)(k0 + ty + i) * DMODEL + n0 + tx];
  __syncthreads();
#pragma unroll
  for (int i = 0; i < 4; ++i)
    out[(size_t)(n0 + ty + i) * DMODEL + k0 + tx] = (bf16)tile[tx][ty + i];
}

// ---------------------------------------------------------------------------
// GEMM: Y[M,768] = X[M,768] @ W + bias, with W supplied TRANSPOSED (Wt[n][k]).
// mode 0: write bf16 head-split [B,H,S,DH]      (Q,K projections)
// mode 2: write bf16 head-split transposed [B,H,DH,S]  (V projection)
// mode 1: write fp32 row-major [M,768]          (output projection -> d_out)
// Block = 256 threads (8 waves), tile 128x128, K-step 32.
// ---------------------------------------------------------------------------
__global__ __launch_bounds__(256)
void gemm_wmma(const bf16* __restrict__ X, const bf16* __restrict__ Wt,
               const float* __restrict__ bias, bf16* __restrict__ outb,
               float* __restrict__ outf, int mode) {
  __shared__ bf16 sA[128][32];   // [m][k]
  __shared__ bf16 sB[128][32];   // [n][k]

  const int tid  = threadIdx.x;
  const int lane = tid & 31;
  const int wave = tid >> 5;
  const int hi   = (lane >> 4) & 1;
  const int lo   = lane & 15;
  const int wm   = wave >> 1;        // 0..3 -> 32-row strip
  const int wn   = wave & 1;         // 0..1 -> 64-col strip
  const int rowBase = blockIdx.x * 128;
  const int colBase = blockIdx.y * 128;

  v8f acc[2][4];
#pragma unroll
  for (int mt = 0; mt < 2; ++mt)
#pragma unroll
    for (int nt = 0; nt < 4; ++nt) acc[mt][nt] = zero8();

  for (int k0 = 0; k0 < DMODEL; k0 += 32) {
    // Stage A and (pre-transposed) B tiles: contiguous 16B chunks only.
#pragma unroll
    for (int c = 0; c < 2; ++c) {
      int chunk = tid * 2 + c;
      int row = chunk >> 2, seg = (chunk & 3) * 8;
      stage16(X + (size_t)(rowBase + row) * DMODEL + k0 + seg, &sA[row][seg]);
      stage16(Wt + (size_t)(colBase + row) * DMODEL + k0 + seg, &sB[row][seg]);
    }
    if (k0 + 32 < DMODEL)
      __builtin_prefetch(X + (size_t)(rowBase + (tid >> 1)) * DMODEL + k0 + 32, 0, 3);
    stage_wait();
    __syncthreads();

    v16bf afrag[2], bfrag[4];
#pragma unroll
    for (int mt = 0; mt < 2; ++mt) {
      int r = wm * 32 + mt * 16 + lo;
      afrag[mt] = ld_frag(&sA[r][hi * 8], &sA[r][16 + hi * 8]);
    }
#pragma unroll
    for (int nt = 0; nt < 4; ++nt) {
      int r = wn * 64 + nt * 16 + lo;
      bfrag[nt] = ld_frag(&sB[r][hi * 8], &sB[r][16 + hi * 8]);
    }
#pragma unroll
    for (int mt = 0; mt < 2; ++mt)
#pragma unroll
      for (int nt = 0; nt < 4; ++nt)
        acc[mt][nt] = wmma_bf16(afrag[mt], bfrag[nt], acc[mt][nt]);
    __syncthreads();
  }

  // Epilogue: bias add + scatter.
#pragma unroll
  for (int mt = 0; mt < 2; ++mt) {
#pragma unroll
    for (int nt = 0; nt < 4; ++nt) {
      int col = colBase + wn * 64 + nt * 16 + lo;
      float bcol = bias[col];
#pragma unroll
      for (int r = 0; r < 8; ++r) {
        int row = rowBase + wm * 32 + mt * 16 + r + 8 * hi;
        float val = acc[mt][nt][r] + bcol;
        int b = row >> 12, s = row & (SEQ - 1);
        int h = col >> 6, dh = col & 63;
        if (mode == 0) {
          outb[(((size_t)(b * HEADS + h)) * SEQ + s) * DHEAD + dh] = (bf16)val;
        } else if (mode == 2) {
          outb[(((size_t)(b * HEADS + h)) * DHEAD + dh) * SEQ + s] = (bf16)val;
        } else {
          outf[(size_t)row * DMODEL + col] = val;
        }
      }
    }
  }
}

// ---------------------------------------------------------------------------
// Flash attention: one block = 128 q-rows of one (batch, head).
// 8 waves x 16 q-rows; loop over 64-key blocks; online softmax.
// Q/K: bf16 [B*H, S, 64]; Vt: bf16 [B*H, 64, S]. Ctx out: bf16 [B*S, 768].
// ---------------------------------------------------------------------------
__global__ __launch_bounds__(256)
void flash_attn(const bf16* __restrict__ Q, const bf16* __restrict__ K,
                const bf16* __restrict__ Vt, bf16* __restrict__ Ctx) {
  __shared__ bf16 sK[64][64];        // [key][dh]  == B-layout for Q*K^T
  __shared__ bf16 sVt[64][64];       // [dh][key]  == B-layout for P*V
  __shared__ bf16 sP[8][16][64];     // per-wave prob staging (C-layout -> A-layout)

  const int tid  = threadIdx.x;
  const int lane = tid & 31;
  const int wave = tid >> 5;
  const int hi   = (lane >> 4) & 1;
  const int lo   = lane & 15;
  const int bh   = blockIdx.y;                 // b*H + h
  const int qBase = blockIdx.x * 128;
  const size_t headOff = (size_t)bh * SEQ * DHEAD;

  // Q fragments: 16 q-rows x 64 dh, kept in registers for the whole kernel.
  v16bf aq[2];
  {
    const bf16* qp = Q + headOff + (size_t)(qBase + wave * 16 + lo) * DHEAD;
#pragma unroll
    for (int c = 0; c < 2; ++c)
      aq[c] = ld_frag(qp + c * 32 + hi * 8, qp + c * 32 + 16 + hi * 8);
  }

  v8f cacc[4];                       // 16 q x 64 dh accumulator
  float mrun[8], lrun[8];
#pragma unroll
  for (int t = 0; t < 4; ++t) cacc[t] = zero8();
#pragma unroll
  for (int r = 0; r < 8; ++r) { mrun[r] = -1e30f; lrun[r] = 0.f; }

  for (int kb = 0; kb < SEQ; kb += 64) {
    // Stage K tile [key][dh] and pre-transposed V tile [dh][key]: all 16B chunks.
#pragma unroll
    for (int c = 0; c < 2; ++c) {
      int chunk = tid * 2 + c;
      int row = chunk >> 3, seg = (chunk & 7) * 8;
      stage16(K + headOff + (size_t)(kb + row) * DHEAD + seg, &sK[row][seg]);
      stage16(Vt + headOff + (size_t)row * SEQ + kb + seg, &sVt[row][seg]);
    }
    stage_wait();
    __syncthreads();

    // Scores: 16 q x 64 keys = 4 tiles, each accumulated over dh (2 x k32).
    v8f sacc[4];
#pragma unroll
    for (int j = 0; j < 4; ++j) sacc[j] = zero8();
#pragma unroll
    for (int c = 0; c < 2; ++c) {
#pragma unroll
      for (int j = 0; j < 4; ++j) {
        const bf16* kp = &sK[j * 16 + lo][c * 32 + hi * 8];
        sacc[j] = wmma_bf16(aq[c], ld_frag(kp, kp + 16), sacc[j]);
      }
    }

    // Online softmax (per q-row r; rows live within a 16-lane half).
#pragma unroll
    for (int r = 0; r < 8; ++r) {
      float mx = sacc[0][r];
#pragma unroll
      for (int j = 1; j < 4; ++j) mx = fmaxf(mx, sacc[j][r]);
#pragma unroll
      for (int off = 1; off < 16; off <<= 1)
        mx = fmaxf(mx, __shfl_xor(mx, off, 32));
      float mnew  = fmaxf(mrun[r], mx * ATTN_SCALE);
      float alpha = __expf(mrun[r] - mnew);
      float rsum  = 0.f;
#pragma unroll
      for (int j = 0; j < 4; ++j) {
        float p = __expf(sacc[j][r] * ATTN_SCALE - mnew);
        sacc[j][r] = p;
        rsum += p;
      }
#pragma unroll
      for (int off = 1; off < 16; off <<= 1)
        rsum += __shfl_xor(rsum, off, 32);
      lrun[r] = lrun[r] * alpha + rsum;
      mrun[r] = mnew;
#pragma unroll
      for (int t = 0; t < 4; ++t) cacc[t][r] *= alpha;
    }

    // Re-swizzle probs C-layout -> A-layout through per-wave LDS.
#pragma unroll
    for (int j = 0; j < 4; ++j)
#pragma unroll
      for (int r = 0; r < 8; ++r)
        sP[wave][r + 8 * hi][j * 16 + lo] = (bf16)sacc[j][r];

    // ctx += P x V  (k = key dim, 2 x 32-chunks).
#pragma unroll
    for (int c = 0; c < 2; ++c) {
      const bf16* pp = &sP[wave][lo][c * 32];
      v16bf ap = ld_frag(pp + hi * 8, pp + 16 + hi * 8);
#pragma unroll
      for (int t = 0; t < 4; ++t) {
        const bf16* vp = &sVt[t * 16 + lo][c * 32 + hi * 8];
        cacc[t] = wmma_bf16(ap, ld_frag(vp, vp + 16), cacc[t]);
      }
    }
    __syncthreads();
  }

  // Normalize and write merged-head bf16 ctx [B*S, 768].
  const int b = bh / HEADS, h = bh % HEADS;
#pragma unroll
  for (int r = 0; r < 8; ++r) {
    float inv = 1.0f / lrun[r];
    int q = qBase + wave * 16 + r + 8 * hi;
    size_t rowOff = ((size_t)b * SEQ + q) * DMODEL + h * DHEAD;
#pragma unroll
    for (int t = 0; t < 4; ++t)
      Ctx[rowOff + t * 16 + lo] = (bf16)(cacc[t][r] * inv);
  }
}

// ---------------------------------------------------------------------------
extern "C" void kernel_launch(void* const* d_in, const int* in_sizes, int n_in,
                              void* d_out, int out_size, void* d_ws, size_t ws_size,
                              hipStream_t stream) {
  const float* hs = (const float*)d_in[0];
  const float* Wq = (const float*)d_in[1];
  const float* bq = (const float*)d_in[2];
  const float* Wk = (const float*)d_in[3];
  const float* bk = (const float*)d_in[4];
  const float* Wv = (const float*)d_in[5];
  const float* bv = (const float*)d_in[6];
  const float* Wo = (const float*)d_in[7];
  const float* bo = (const float*)d_in[8];
  float* out = (float*)d_out;

  char* ws = (char*)d_ws;
  size_t off = 0;
  auto alloc_bf16 = [&](size_t elems) {
    bf16* p = (bf16*)(ws + off);
    off += elems * sizeof(bf16);
    return p;
  };
  bf16* Xb  = alloc_bf16((size_t)MROWS * DMODEL);   // hidden states, bf16
  bf16* Wqt = alloc_bf16((size_t)DMODEL * DMODEL);  // transposed bf16 weights
  bf16* Wkt = alloc_bf16((size_t)DMODEL * DMODEL);
  bf16* Wvt = alloc_bf16((size_t)DMODEL * DMODEL);
  bf16* Wot = alloc_bf16((size_t)DMODEL * DMODEL);
  bf16* Qh  = alloc_bf16((size_t)MROWS * DMODEL);   // [B,H,S,DH]
  bf16* Kh  = alloc_bf16((size_t)MROWS * DMODEL);   // [B,H,S,DH]
  bf16* Vht = alloc_bf16((size_t)MROWS * DMODEL);   // [B,H,DH,S]
  bf16* Cb  = alloc_bf16((size_t)MROWS * DMODEL);   // merged-head ctx

  const int nX = MROWS * DMODEL;
  f32_to_bf16<<<(nX + 255) / 256, 256, 0, stream>>>(hs, Xb, nX);
  dim3 tg(DMODEL / 32, DMODEL / 32);    // 24 x 24
  transpose_f32_to_bf16<<<tg, 256, 0, stream>>>(Wq, Wqt);
  transpose_f32_to_bf16<<<tg, 256, 0, stream>>>(Wk, Wkt);
  transpose_f32_to_bf16<<<tg, 256, 0, stream>>>(Wv, Wvt);
  transpose_f32_to_bf16<<<tg, 256, 0, stream>>>(Wo, Wot);

  dim3 gg(MROWS / 128, DMODEL / 128);   // 64 x 6
  gemm_wmma<<<gg, 256, 0, stream>>>(Xb, Wqt, bq, Qh, nullptr, 0);
  gemm_wmma<<<gg, 256, 0, stream>>>(Xb, Wkt, bk, Kh, nullptr, 0);
  gemm_wmma<<<gg, 256, 0, stream>>>(Xb, Wvt, bv, Vht, nullptr, 2);

  dim3 fg(SEQ / 128, BATCH * HEADS);    // 32 x 24
  flash_attn<<<fg, 256, 0, stream>>>(Qh, Kh, Vht, Cb);

  gemm_wmma<<<gg, 256, 0, stream>>>(Cb, Wot, bo, nullptr, out, 1);
}
